// QuantizedLinearLayer_66082366816859
// MI455X (gfx1250) — matrix operand order repaired and checked by reading, not compiled, not run on hardware
//
#include <hip/hip_runtime.h>
#include <stdint.h>

// GEMM dims: M = B*S = 8192, N = D_OUT = 11008, K = D_IN = 4096
#define M_DIM 8192
#define N_DIM 11008
#define K_DIM 4096

// Block tile
#define BM 128
#define BN 128
#define BK 64
// LDS row strides (in f16 elements), padded +8 to avoid bank conflicts
#define LDA 72
#define LDB 72

typedef __attribute__((ext_vector_type(16))) _Float16 v16h;
typedef __attribute__((ext_vector_type(8)))  _Float16 v8h;
typedef __attribute__((ext_vector_type(4)))  _Float16 v4h;
typedef __attribute__((ext_vector_type(8)))  float    v8f;
typedef __attribute__((ext_vector_type(4)))  float    v4f;
typedef __attribute__((ext_vector_type(4)))  int      v4i;

__global__ __launch_bounds__(256)
void QuantizedLinearLayer_66082366816859_kernel(const float* __restrict__ A,
                                                const int8_t* __restrict__ W,
                                                const float* __restrict__ scale,
                                                const float* __restrict__ bias,
                                                float* __restrict__ Out)
{
    __shared__ _Float16 As[BM * LDA];  // 128 x 64 (+pad) f16 activations
    __shared__ _Float16 Bs[BN * LDB];  // 128 x 64 (+pad) f16 weights (N-major)

    const int tid  = threadIdx.x;
    const int lane = tid & 31;          // wave32
    const int wave = tid >> 5;          // 8 waves

    const int blockN = blockIdx.x * BN;
    const int blockM = blockIdx.y * BM;

    // 2 waves tile M (64 rows each), 4 waves tile N (32 cols each)
    const int waveM = (wave & 1) * 64;
    const int waveN = (wave >> 1) * 32;

    const int l15    = lane & 15;
    const int hihalf = lane >> 4;       // 0 for lanes 0-15, 1 for lanes 16-31

    v8f acc[4][2] = {};                 // 4 M-tiles x 2 N-tiles of 16x16 f32

    for (int k0 = 0; k0 < K_DIM; k0 += BK) {
        // ---- Stage A tile: 128 rows x 64 f32 -> f16 LDS (8 x b128 loads/thread)
        #pragma unroll
        for (int i = 0; i < 8; ++i) {
            int c   = tid + i * 256;          // 2048 float4 chunks
            int row = c >> 4;                 // 16 float4 per row
            int col = (c & 15) << 2;          // f32 column
            v4f v = *(const v4f*)(A + (size_t)(blockM + row) * K_DIM + k0 + col);
            v4h h;
            h.x = (_Float16)v.x; h.y = (_Float16)v.y;
            h.z = (_Float16)v.z; h.w = (_Float16)v.w;
            *(v4h*)(&As[row * LDA + col]) = h;
        }
        // ---- Stage W tile: 128 N-rows x 64 int8 -> f16 LDS (2 x b128 loads/thread)
        #pragma unroll
        for (int i = 0; i < 2; ++i) {
            int c   = tid + i * 256;          // 512 16-byte chunks
            int row = c >> 2;                 // 4 chunks per row
            int col = (c & 3) << 4;           // int8 column
            v4i w = *(const v4i*)(W + (size_t)(blockN + row) * K_DIM + k0 + col);
            #pragma unroll
            for (int q = 0; q < 4; ++q) {
                int d = ((const int*)&w)[q];
                v4h h;
                h.x = (_Float16)(float)(int8_t)(d);
                h.y = (_Float16)(float)(int8_t)(d >> 8);
                h.z = (_Float16)(float)(int8_t)(d >> 16);
                h.w = (_Float16)(float)(int8_t)(d >> 24);
                *(v4h*)(&Bs[row * LDB + col + q * 4]) = h;
            }
        }
        __syncthreads();

        // Prefetch next K-tile into cache (global_prefetch_b8)
        if (k0 + BK < K_DIM) {
            __builtin_prefetch(A + (size_t)(blockM + (tid >> 1)) * K_DIM + k0 + BK, 0, 3);
            __builtin_prefetch(W + (size_t)(blockN + (tid >> 1)) * K_DIM + k0 + BK, 0, 3);
        }

        // ---- Two K=32 WMMA steps over the staged K=64
        #pragma unroll
        for (int kk = 0; kk < BK; kk += 32) {
            // A fragment (16x32 f16): lanes 0-15 hold K 0-7 & 16-23,
            // lanes 16-31 hold K 8-15 & 24-31 (ISA 16-bit A layout)
            v16h afrag[4];
            #pragma unroll
            for (int i = 0; i < 4; ++i) {
                const _Float16* p = &As[(waveM + i * 16 + l15) * LDA + kk + (hihalf << 3)];
                union { v16h v; v8h h[2]; } u;
                u.h[0] = *(const v8h*)(p);
                u.h[1] = *(const v8h*)(p + 16);
                afrag[i] = u.v;
            }
            // B fragment (32x16 f16): column N = lane%16; lanes 0-15 hold K 0-7 & 16-23,
            // lanes 16-31 hold K 8-15 & 24-31 (same order as A)
            v16h bfrag[2];
            #pragma unroll
            for (int j = 0; j < 2; ++j) {
                const _Float16* p = &Bs[(waveN + j * 16 + l15) * LDB + kk + (hihalf << 3)];
                union { v16h v; v8h h[2]; } u;
                u.h[0] = *(const v8h*)(p);
                u.h[1] = *(const v8h*)(p + 16);
                bfrag[j] = u.v;
            }
            #pragma unroll
            for (int i = 0; i < 4; ++i)
                #pragma unroll
                for (int j = 0; j < 2; ++j)
                    acc[i][j] = __builtin_amdgcn_wmma_f32_16x16x32_f16(
                        false, afrag[i], false, bfrag[j],
                        (short)0, acc[i][j], false, false);
        }
        __syncthreads();
    }

    // ---- Epilogue: C layout is VGPR r -> M = r (lanes 0-15) / M = r+8 (lanes 16-31),
    // N = lane%16. Apply per-channel scale + bias in f32.
    const int rowSel = hihalf << 3;
    #pragma unroll
    for (int j = 0; j < 2; ++j) {
        const int n = blockN + waveN + j * 16 + l15;
        const float s  = scale[n];
        const float bb = bias[n];
        #pragma unroll
        for (int i = 0; i < 4; ++i) {
            const int mbase = blockM + waveM + i * 16 + rowSel;
            #pragma unroll
            for (int r = 0; r < 8; ++r) {
                Out[(size_t)(mbase + r) * N_DIM + n] = acc[i][j][r] * s + bb;
            }
        }
    }
}

extern "C" void kernel_launch(void* const* d_in, const int* in_sizes, int n_in,
                              void* d_out, int out_size, void* d_ws, size_t ws_size,
                              hipStream_t stream) {
    const float*  A     = (const float*)d_in[0];   // [B,S,D_IN] f32
    const int8_t* W     = (const int8_t*)d_in[1];  // [D_OUT,D_IN] int8
    const float*  scale = (const float*)d_in[2];   // [D_OUT]
    const float*  bias  = (const float*)d_in[3];   // [1,D_OUT]
    float*        Out   = (float*)d_out;           // [B,S,D_OUT] f32

    dim3 grid(N_DIM / BN, M_DIM / BM);  // (86, 64)
    dim3 block(256);
    QuantizedLinearLayer_66082366816859_kernel<<<grid, block, 0, stream>>>(
        A, W, scale, bias, Out);
}
